// LSTMModel_16621523435787
// MI455X (gfx1250) — matrix operand order repaired and checked by reading, not compiled, not run on hardware
//
#include <hip/hip_runtime.h>

#define HID   50
#define NPACK 150   // packed rows: [i(50); g(50); o(50)]  (f-gate is dead)
#define NT    10    // 16-wide N tiles covering 150 (+10 pad) rows

typedef __attribute__((ext_vector_type(16))) _Float16 v16h;
typedef __attribute__((ext_vector_type(8)))  float    v8f;

// gfx1250 has a native v_tanh_f32 TRANS op; use it if the toolchain exposes it.
#if __has_builtin(__builtin_amdgcn_tanhf)
static __device__ __forceinline__ float fast_tanh(float x) {
  return __builtin_amdgcn_tanhf(x);
}
static __device__ __forceinline__ float fast_sigmoid(float x) {
  return fmaf(0.5f, __builtin_amdgcn_tanhf(0.5f * x), 0.5f);
}
#else
static __device__ __forceinline__ float fast_tanh(float x) {
  float e = __builtin_amdgcn_exp2f(2.8853900817779268f * x);    // exp(2x)
  return 1.0f - 2.0f * __builtin_amdgcn_rcpf(e + 1.0f);
}
static __device__ __forceinline__ float fast_sigmoid(float x) {
  float e = __builtin_amdgcn_exp2f(-1.4426950408889634f * x);   // exp(-x)
  return __builtin_amdgcn_rcpf(1.0f + e);
}
#endif

union IPack { int i; _Float16 h[2]; };
union AtoV  { int i[8]; v16h v; };

__global__ __launch_bounds__(32, 1)
void lstm_autoreg_kernel(const float* __restrict__ batch,
                         const float* __restrict__ Wih0,
                         const float* __restrict__ bih0,
                         const float* __restrict__ bhh0,
                         const float* __restrict__ Wih1,
                         const float* __restrict__ bih1,
                         const float* __restrict__ bhh1,
                         const float* __restrict__ Wfc,
                         const float* __restrict__ bfc,
                         float* __restrict__ out,
                         float* __restrict__ dump,
                         int T)
{
  const int lane = threadIdx.x;      // 0..31, wave32
  const int col  = lane & 15;
  const int hi16 = lane >> 4;

  // ---- layer-0 per-lane constants: lane L owns hidden units k=2L, 2L+1.
  // Lanes >= 25 get zero weights/biases -> their h0 pair is exactly 0.
  const int k0 = 2 * lane, k1 = 2 * lane + 1;
  const bool ok0 = (k0 < HID), ok1 = (k1 < HID);
  const float wi0 = ok0 ? Wih0[k0]           : 0.f;
  const float wi1 = ok1 ? Wih0[k1]           : 0.f;
  const float wg0 = ok0 ? Wih0[2 * HID + k0] : 0.f;
  const float wg1 = ok1 ? Wih0[2 * HID + k1] : 0.f;
  const float wo0 = ok0 ? Wih0[3 * HID + k0] : 0.f;
  const float wo1 = ok1 ? Wih0[3 * HID + k1] : 0.f;
  const float bi0 = ok0 ? bih0[k0] + bhh0[k0]                     : 0.f;
  const float bi1 = ok1 ? bih0[k1] + bhh0[k1]                     : 0.f;
  const float bg0 = ok0 ? bih0[2 * HID + k0] + bhh0[2 * HID + k0] : 0.f;
  const float bg1 = ok1 ? bih0[2 * HID + k1] + bhh0[2 * HID + k1] : 0.f;
  const float bo0 = ok0 ? bih0[3 * HID + k0] + bhh0[3 * HID + k0] : 0.f;
  const float bo1 = ok1 ? bih0[3 * HID + k1] + bhh0[3 * HID + k1] : 0.f;
  const float fc0 = ok0 ? Wfc[k0] : 0.f;
  const float fc1 = ok1 ? Wfc[k1] : 0.f;
  const float bfc0 = bfc[0];

  // Bias injection: K=50 of the A vector is forced to 1.0 (lane 25, pair elem 0)
  // so that B row K=50 can carry the layer-1 combined bias -> no post-WMMA add.
  const float selA = (lane == 25) ? 0.f : 1.f;
  const float selB = (lane == 25) ? 1.f : 0.f;

  // ---- layer-1 weights as WMMA B tiles (held in VGPRs for the whole run).
  // Packed col n (0..149): i rows 0..49 -> Wih1 row n; g/o -> row n+50.
  // B packing used here (contiguous k, matches A): VGPR j holds rows K = kt*32 + hi16*16 + 2j (+1),
  // column = lane&15.  Row K==50 carries the bias (multiplied by A[50]==1).
  v16h Bt[NT][2];
#pragma unroll
  for (int t = 0; t < NT; ++t) {
    const int n = 16 * t + col;
    const int r = n + (n >= HID ? HID : 0);
    const bool nv = (n < NPACK);
#pragma unroll
    for (int kt = 0; kt < 2; ++kt) {
      v16h b = {};
#pragma unroll
      for (int j = 0; j < 8; ++j) {
        const int kk = kt * 32 + hi16 * 16 + 2 * j;   // even
        float lo;
        if (nv && kk < HID)        lo = Wih1[r * HID + kk];
        else if (nv && kk == HID)  lo = bih1[r] + bhh1[r];   // bias row
        else                       lo = 0.f;
        const float hi = (nv && kk + 1 < HID) ? Wih1[r * HID + kk + 1] : 0.f;
        b[2 * j]     = (_Float16)lo;
        b[2 * j + 1] = (_Float16)hi;
      }
      Bt[t][kt] = b;
    }
  }

  // ---- ds_bpermute byte-indices to scatter h0 f16-pairs into the A layout.
  // A row 0 as packed here (contiguous k, matches B): lane0 carries K0-15; lane16 K16-31.
  int idxA0[8], idxA1[8];
#pragma unroll
  for (int j = 0; j < 8; ++j) {
    const int p = j + 8 * hi16;             // f16-pair index for VGPR j
    idxA0[j] = 4 * p;                        // K tile 0 (K=0..31)
    idxA1[j] = 4 * (p + 16);                 // K tile 1 (K=32..63)
  }

  // ---- gate-gather offsets into the g1 LDS buffer (tile t -> 32 floats,
  // valid data in the first 16 = D-matrix row 0, lanes 0-15).
  __shared__ float g1buf[11 * 32];
  auto offs = [](int row) { return (row >> 4) * 32 + (row & 15); };
  const int oi0 = offs(2 * lane),           oi1 = offs(2 * lane + 1);
  const int og0 = offs(HID + 2 * lane),     og1 = offs(HID + 2 * lane + 1);
  const int oo0 = offs(2 * HID + 2 * lane), oo1 = offs(2 * HID + 2 * lane + 1);

  // Output pointer: lane 0 walks the real output, others overwrite a dump slot.
  float* outp = (lane == 0) ? out : (dump + lane);
  const int outstep = (lane == 0) ? 1 : 0;

  float x = batch[0];   // only batch element ever used (autonomous mode)

  for (int t = 0; t < T; ++t) {
    // ---- layer 0: c = sig(i)*tanh(g); h = sig(o)*tanh(c)
    const float pi0 = fmaf(x, wi0, bi0), pi1 = fmaf(x, wi1, bi1);
    const float pg0 = fmaf(x, wg0, bg0), pg1 = fmaf(x, wg1, bg1);
    const float po0 = fmaf(x, wo0, bo0), po1 = fmaf(x, wo1, bo1);
    float h0a = fast_sigmoid(po0) * fast_tanh(fast_sigmoid(pi0) * fast_tanh(pg0));
    const float h0b = fast_sigmoid(po1) * fast_tanh(fast_sigmoid(pi1) * fast_tanh(pg1));
    h0a = fmaf(h0a, selA, selB);          // lane 25: h0a := 1.0 (bias input)
    IPack up; up.h[0] = (_Float16)h0a; up.h[1] = (_Float16)h0b;
    const int hpair = up.i;

    // ---- build A tiles (row 0 = h0 | 1.0 bias slot; rows 1-15 finite dupes)
    AtoV a0, a1;
#pragma unroll
    for (int j = 0; j < 8; ++j) {
      a0.i[j] = __builtin_amdgcn_ds_bpermute(idxA0[j], hpair);
      a1.i[j] = __builtin_amdgcn_ds_bpermute(idxA1[j], hpair);
    }

    // ---- layer-1 matvec+bias: 10 N-tiles x 2 chained K-tiles of WMMA f16.
    // D row 0 (acc[0], lanes 0-15) is complete -> straight to LDS, no VALU
    // touch of the accumulator (avoids the WMMA->VALU 4-NOP hazard).
#pragma unroll
    for (int tt = 0; tt < NT; ++tt) {
      v8f zero = {0.f, 0.f, 0.f, 0.f, 0.f, 0.f, 0.f, 0.f};
      v8f acc = __builtin_amdgcn_wmma_f32_16x16x32_f16(
          false, a0.v, false, Bt[tt][0], (short)0, zero, false, false);
      acc = __builtin_amdgcn_wmma_f32_16x16x32_f16(
          false, a1.v, false, Bt[tt][1], (short)0, acc, false, false);
      g1buf[tt * 32 + lane] = acc[0];
    }
    __syncthreads();   // single-wave WG: barrier is free, acts as fence

    // ---- layer-1 gates + fc partial (lanes >= 25 have fc weights == 0)
    const float qi0 = g1buf[oi0], qi1 = g1buf[oi1];
    const float qg0 = g1buf[og0], qg1 = g1buf[og1];
    const float qo0 = g1buf[oo0], qo1 = g1buf[oo1];
    const float e0 = fast_sigmoid(qo0) * fast_tanh(fast_sigmoid(qi0) * fast_tanh(qg0));
    const float e1 = fast_sigmoid(qo1) * fast_tanh(fast_sigmoid(qi1) * fast_tanh(qg1));
    float rsum = fmaf(fmaxf(e0, 0.f), fc0, fmaxf(e1, 0.f) * fc1);

    // ---- wave32 xor-reduction: every lane ends with the full sum
#pragma unroll
    for (int off = 16; off; off >>= 1) rsum += __shfl_xor(rsum, off, 32);
    const float pred = rsum + bfc0;

    *outp = pred;
    outp += outstep;

    x = pred;            // autoregressive feedback
    // no trailing barrier: in-wave DS ordering covers next iter's stores (WAR)
  }
}

extern "C" void kernel_launch(void* const* d_in, const int* in_sizes, int n_in,
                              void* d_out, int out_size, void* d_ws, size_t ws_size,
                              hipStream_t stream) {
  const float* batch = (const float*)d_in[0];
  const float* Wih0  = (const float*)d_in[1];
  // d_in[2] = Whh0 (dead: h is always 0)
  const float* bih0  = (const float*)d_in[3];
  const float* bhh0  = (const float*)d_in[4];
  const float* Wih1  = (const float*)d_in[5];
  // d_in[6] = Whh1 (dead)
  const float* bih1  = (const float*)d_in[7];
  const float* bhh1  = (const float*)d_in[8];
  const float* Wfc   = (const float*)d_in[9];
  const float* bfc   = (const float*)d_in[10];

  lstm_autoreg_kernel<<<1, 32, 0, stream>>>(
      batch, Wih0, bih0, bhh0, Wih1, bih1, bhh1, Wfc, bfc,
      (float*)d_out, (float*)d_ws, out_size);
}
